// SF_Gatv2_Layers_69054484185858
// MI455X (gfx1250) — compile-verified, matrix operand order checked by reading, not run on hardware
//
#include <hip/hip_runtime.h>
#include <hip/hip_bf16.h>

typedef __attribute__((ext_vector_type(16))) __bf16 v16bf;
typedef __attribute__((ext_vector_type(8)))  __bf16 v8bf;
typedef __attribute__((ext_vector_type(8)))  float  v8f;

#define N_NODES  20000
#define N_EDGES  640000
#define N_GRAPHS 64
#define HEADS    4
#define OUTDIM   128
#define HC_MAX   512

// ---------- helpers ----------
__device__ __forceinline__ void atomicMaxF(float* addr, float v) {
  if (v >= 0.0f) atomicMax((int*)addr, __float_as_int(v));
  else           atomicMin((unsigned int*)addr, (unsigned int)__float_as_int(v));
}

__global__ void k_init_f32(float* __restrict__ p, float v, int n) {
  int i = blockIdx.x * blockDim.x + threadIdx.x;
  if (i < n) p[i] = v;
}

__global__ void k_f32_to_bf16(const float* __restrict__ s, __bf16* __restrict__ d, int n) {
  int i = blockIdx.x * blockDim.x + threadIdx.x;
  if (i < n) d[i] = (__bf16)s[i];
}

// W: [K, Nn] row-major fp32  ->  Wt: [Nn, K] bf16
__global__ void k_w_transpose_bf16(const float* __restrict__ W, __bf16* __restrict__ Wt,
                                   int K, int Nn) {
  int i = blockIdx.x * blockDim.x + threadIdx.x;
  if (i < K * Nn) {
    int n = i / K, k = i % K;
    Wt[i] = (__bf16)W[k * Nn + n];
  }
}

// ---------- WMMA GEMM: C[M,Nn] = A[M,K] * Bt[Nn,K]^T + bias ----------
// one wave (32 lanes) per 16x16 output tile, K stepped by 32 (bf16 WMMA)
__global__ void k_gemm_wmma_bf16(const __bf16* __restrict__ A,
                                 const __bf16* __restrict__ Bt,
                                 const float*  __restrict__ bias,
                                 float*        __restrict__ Cmat,
                                 int M, int K, int Nn) {
  const int lane = threadIdx.x;        // 0..31
  const int half = lane >> 4;          // 0 or 1
  const int l16  = lane & 15;
  const int tm = blockIdx.y, tn = blockIdx.x;
  const int rowA = tm * 16 + l16;      // M index this lane loads
  const int rowB = tn * 16 + l16;      // N index this lane loads (Bt row)
  const __bf16* Abase = A  + (size_t)rowA * K + half * 8;
  const __bf16* Bbase = Bt + (size_t)rowB * K + half * 8;

  v8f acc = {};
  for (int kk = 0; kk < K; kk += 32) {
    v8bf alo = *(const v8bf*)(Abase + kk);
    v8bf ahi = *(const v8bf*)(Abase + kk + 16);
    v8bf blo = *(const v8bf*)(Bbase + kk);
    v8bf bhi = *(const v8bf*)(Bbase + kk + 16);
    v16bf av, bv;
#pragma unroll
    for (int i = 0; i < 8; ++i) {
      av[i] = alo[i]; av[i + 8] = ahi[i];
      bv[i] = blo[i]; bv[i + 8] = bhi[i];
    }
    acc = __builtin_amdgcn_wmma_f32_16x16x32_bf16(
        /*neg_a=*/false, av, /*neg_b=*/false, bv,
        /*c_mod=*/(short)0, acc, /*reuse_a=*/false, /*reuse_b=*/false);
  }

  const int nIdx = tn * 16 + l16;
  const float b = bias[nIdx];
#pragma unroll
  for (int r = 0; r < 8; ++r) {
    const int m = tm * 16 + r + half * 8;   // f32 C/D layout: VGPR r, half selects M+8
    Cmat[(size_t)m * Nn + nIdx] = acc[r] + b;
  }
}

// ---------- edge pass 1: logits + per-dst atomic max ----------
// one wave per 8 consecutive edges (64 per block); We (16 x hc) and att (hc) staged in LDS; e computed on the fly
__global__ void k_edge_logits(const float* __restrict__ xl, const float* __restrict__ xr,
                              const float* __restrict__ edge_attr,
                              const int*   __restrict__ ei,
                              const float* __restrict__ We,   // [16, hc]
                              const float* __restrict__ att,  // [hc] = [H,C]
                              float* __restrict__ logits,     // [E, H]
                              float* __restrict__ nmax,       // [N, H]
                              int hc, int C) {
  __shared__ float sWe[16 * HC_MAX];
  __shared__ float sAtt[HC_MAX];
  const int tid = threadIdx.x;
  for (int i = tid; i < 16 * hc; i += blockDim.x) sWe[i] = We[i];
  for (int i = tid; i < hc;      i += blockDim.x) sAtt[i] = att[i];
  __syncthreads();

  const int wave = tid >> 5;
  const int lane = tid & 31;
  const int base = blockIdx.x * 64 + wave * 8;

  for (int it = 0; it < 8; ++it) {
    const int e = base + it;
    if (e >= N_EDGES) break;
    const int src = ei[e], dst = ei[N_EDGES + e];
    float ea[16];
#pragma unroll
    for (int j = 0; j < 16; ++j) ea[j] = edge_attr[(size_t)e * 16 + j];

#pragma unroll
    for (int h = 0; h < HEADS; ++h) {
      float a = 0.f;
      for (int ch = h * C + lane; ch < (h + 1) * C; ch += 32) {
        float ev = 0.f;
#pragma unroll
        for (int j = 0; j < 16; ++j) ev += ea[j] * sWe[j * hc + ch];
        float v = xl[(size_t)src * hc + ch] + xr[(size_t)dst * hc + ch] + ev;
        v = (v > 0.f) ? v : 0.2f * v;           // leaky relu
        a += v * sAtt[ch];
      }
      for (int off = 16; off > 0; off >>= 1) a += __shfl_xor(a, off, 32);
      if (lane == 0) {
        logits[(size_t)e * HEADS + h] = a;
        atomicMaxF(&nmax[(size_t)dst * HEADS + h], a);
      }
    }
  }
}

// ---------- edge pass 2: p = exp(logit - max), denom += p ----------
__global__ void k_edge_exp(float* __restrict__ lp, const int* __restrict__ ei,
                           const float* __restrict__ nmax, float* __restrict__ denom) {
  int t = blockIdx.x * blockDim.x + threadIdx.x;
  if (t < N_EDGES * HEADS) {
    const int e = t >> 2, h = t & 3;
    const int dst = ei[N_EDGES + e];
    const float p = expf(lp[t] - nmax[(size_t)dst * HEADS + h]);
    lp[t] = p;
    atomicAdd(&denom[(size_t)dst * HEADS + h], p);
  }
}

// ---------- edge pass 3: out[dst] += alpha * xl[src] ----------
__global__ void k_edge_agg(const float* __restrict__ p, const float* __restrict__ denom,
                           const float* __restrict__ xl, const int* __restrict__ ei,
                           float* __restrict__ out, int hc, int C) {
  const long long t = (long long)blockIdx.x * blockDim.x + threadIdx.x;
  const long long tot = (long long)N_EDGES * hc;
  if (t < tot) {
    const int e  = (int)(t / hc);
    const int ch = (int)(t % hc);
    const int h  = ch / C;
    const int src = ei[e], dst = ei[N_EDGES + e];
    const float alpha = p[(size_t)e * HEADS + h] /
                        (denom[(size_t)dst * HEADS + h] + 1e-16f);
    atomicAdd(&out[(size_t)dst * hc + ch], alpha * xl[(size_t)src * hc + ch]);
  }
}

// ---------- layer finalize ----------
__global__ void k_finalize_concat(const float* __restrict__ acc, const float* __restrict__ bias,
                                  float* __restrict__ h, int hc) {
  int t = blockIdx.x * blockDim.x + threadIdx.x;
  if (t < N_NODES * hc) h[t] = acc[t] + bias[t % hc];
}

__global__ void k_finalize_mean(const float* __restrict__ acc, const float* __restrict__ bias,
                                float* __restrict__ h) {
  int t = blockIdx.x * blockDim.x + threadIdx.x;
  if (t < N_NODES * OUTDIM) {
    const int n = t / OUTDIM, c = t % OUTDIM;
    float s = 0.f;
#pragma unroll
    for (int hh = 0; hh < HEADS; ++hh) s += acc[(size_t)n * (HEADS * OUTDIM) + hh * OUTDIM + c];
    h[t] = s * 0.25f + bias[c];
  }
}

// ---------- graph pooling ----------
__global__ void k_pool(const float* __restrict__ h, const int* __restrict__ batch,
                       float* __restrict__ gmax, float* __restrict__ gsum,
                       float* __restrict__ gcnt) {
  int t = blockIdx.x * blockDim.x + threadIdx.x;
  if (t < N_NODES * OUTDIM) {
    const int n = t / OUTDIM, c = t % OUTDIM;
    const int g = batch[n];
    const float v = h[t];
    atomicMaxF(&gmax[(size_t)g * OUTDIM + c], v);
    atomicAdd(&gsum[(size_t)g * OUTDIM + c], v);
    if (c == 0) atomicAdd(&gcnt[g], 1.0f);
  }
}

__global__ void k_pool_fin(const float* __restrict__ gmax, const float* __restrict__ gsum,
                           const float* __restrict__ gcnt, float* __restrict__ out) {
  int t = blockIdx.x * blockDim.x + threadIdx.x;
  if (t < N_GRAPHS * OUTDIM) {
    const int g = t / OUTDIM, c = t % OUTDIM;
    out[(size_t)g * 2 * OUTDIM + c] = gmax[t];
    out[(size_t)g * 2 * OUTDIM + OUTDIM + c] = gsum[t] / fmaxf(gcnt[g], 1.0f);
  }
}

// ---------- host launch ----------
extern "C" void kernel_launch(void* const* d_in, const int* in_sizes, int n_in,
                              void* d_out, int out_size, void* d_ws, size_t ws_size,
                              hipStream_t stream) {
  (void)in_sizes; (void)n_in; (void)out_size; (void)ws_size;
  const float* x         = (const float*)d_in[0];
  const float* edge_attr = (const float*)d_in[1];
  const int*   ei        = (const int*)d_in[2];
  const int*   batch     = (const int*)d_in[3];

  const float *Wl[3], *bl[3], *Wr[3], *br[3], *We[3], *att[3], *bias[3];
  for (int l = 0; l < 3; ++l) {
    const int b = 4 + 7 * l;
    Wl[l]   = (const float*)d_in[b + 0];
    bl[l]   = (const float*)d_in[b + 1];
    Wr[l]   = (const float*)d_in[b + 2];
    br[l]   = (const float*)d_in[b + 3];
    We[l]   = (const float*)d_in[b + 4];
    att[l]  = (const float*)d_in[b + 5];
    bias[l] = (const float*)d_in[b + 6];
  }
  const int cin_[3] = {128, 256, 256};
  const int C_[3]   = {64, 64, 128};
  const int hc_[3]  = {256, 256, 512};
  const int cc_[3]  = {1, 1, 0};

  char* w = (char*)d_ws;
  size_t off = 0;
  auto carve = [&](size_t bytes) -> void* {
    void* p = w + off;
    off += (bytes + 255) & ~(size_t)255;
    return p;
  };
  float*  hA   = (float*) carve((size_t)N_NODES * 256 * 4);
  float*  hB   = (float*) carve((size_t)N_NODES * 256 * 4);
  __bf16* hbf  = (__bf16*)carve((size_t)N_NODES * 256 * 2);
  float*  xl   = (float*) carve((size_t)N_NODES * 512 * 4);
  float*  xr   = (float*) carve((size_t)N_NODES * 512 * 4);
  float*  oacc = (float*) carve((size_t)N_NODES * 512 * 4);
  __bf16* wtl  = (__bf16*)carve((size_t)512 * 256 * 2);
  __bf16* wtr  = (__bf16*)carve((size_t)512 * 256 * 2);
  float*  lp   = (float*) carve((size_t)N_EDGES * HEADS * 4);
  float*  nmx  = (float*) carve((size_t)N_NODES * HEADS * 4);
  float*  den  = (float*) carve((size_t)N_NODES * HEADS * 4);
  float*  gmax = (float*) carve((size_t)N_GRAPHS * OUTDIM * 4);
  float*  gsum = (float*) carve((size_t)N_GRAPHS * OUTDIM * 4);
  float*  gcnt = (float*) carve((size_t)N_GRAPHS * 4);

  const float NEG_INF = -__builtin_huge_valf();
  const float* hin = x;
  float* houts[3] = { hA, hB, hA };

  for (int l = 0; l < 3; ++l) {
    const int cin = cin_[l], C = C_[l], hc = hc_[l];

    const int nconv = N_NODES * cin;
    k_f32_to_bf16<<<(nconv + 255) / 256, 256, 0, stream>>>(hin, hbf, nconv);
    const int wk = cin * hc;
    k_w_transpose_bf16<<<(wk + 255) / 256, 256, 0, stream>>>(Wl[l], wtl, cin, hc);
    k_w_transpose_bf16<<<(wk + 255) / 256, 256, 0, stream>>>(Wr[l], wtr, cin, hc);

    dim3 gg(hc / 16, N_NODES / 16);
    k_gemm_wmma_bf16<<<gg, 32, 0, stream>>>(hbf, wtl, bl[l], xl, N_NODES, cin, hc);
    k_gemm_wmma_bf16<<<gg, 32, 0, stream>>>(hbf, wtr, br[l], xr, N_NODES, cin, hc);

    k_init_f32<<<(N_NODES * HEADS + 255) / 256, 256, 0, stream>>>(nmx, NEG_INF, N_NODES * HEADS);
    k_init_f32<<<(N_NODES * HEADS + 255) / 256, 256, 0, stream>>>(den, 0.f, N_NODES * HEADS);
    k_init_f32<<<(N_NODES * hc + 255) / 256, 256, 0, stream>>>(oacc, 0.f, N_NODES * hc);

    k_edge_logits<<<N_EDGES / 64, 256, 0, stream>>>(xl, xr, edge_attr, ei, We[l], att[l],
                                                    lp, nmx, hc, C);
    k_edge_exp<<<(N_EDGES * HEADS + 255) / 256, 256, 0, stream>>>(lp, ei, nmx, den);

    const long long tot = (long long)N_EDGES * hc;
    k_edge_agg<<<(unsigned)((tot + 255) / 256), 256, 0, stream>>>(lp, den, xl, ei, oacc, hc, C);

    if (cc_[l])
      k_finalize_concat<<<(N_NODES * hc + 255) / 256, 256, 0, stream>>>(oacc, bias[l], houts[l], hc);
    else
      k_finalize_mean<<<(N_NODES * OUTDIM + 255) / 256, 256, 0, stream>>>(oacc, bias[l], houts[l]);
    hin = houts[l];
  }

  // graph pooling: max + mean per graph
  k_init_f32<<<(N_GRAPHS * OUTDIM + 255) / 256, 256, 0, stream>>>(gmax, NEG_INF, N_GRAPHS * OUTDIM);
  k_init_f32<<<(N_GRAPHS * OUTDIM + 255) / 256, 256, 0, stream>>>(gsum, 0.f, N_GRAPHS * OUTDIM);
  k_init_f32<<<1, 64, 0, stream>>>(gcnt, 0.f, N_GRAPHS);
  k_pool<<<(N_NODES * OUTDIM + 255) / 256, 256, 0, stream>>>(hin, batch, gmax, gsum, gcnt);
  k_pool_fin<<<(N_GRAPHS * OUTDIM + 255) / 256, 256, 0, stream>>>(gmax, gsum, gcnt, (float*)d_out);
}
